// DeepJMTModel_25683904430201
// MI455X (gfx1250) — matrix operand order repaired and checked by reading, not compiled, not run on hardware
//
#include <hip/hip_runtime.h>
#include <hip/hip_bf16.h>
#include <math.h>

// ---------------------------------------------------------------------------
// DeepJMT forward on gfx1250 (wave32, WMMA, TDM).
// Dominant cost: two streaming passes over the dense 8192x8192 adjacency
// (2 x 256MB ~ 22us at 23.3 TB/s). adj tiles move via tensor_load_to_lds
// (double-buffered, TENSORcnt), attn@Wh runs on v_wmma_f32_16x16x32_f16.
// Inner loop: 4x ds_load_b128 (adj) + 4x global b128 (e2/Wh col-major) +
// branchless exp chain + one WMMA.
// ---------------------------------------------------------------------------

typedef __attribute__((ext_vector_type(16))) _Float16 v16h;
typedef __attribute__((ext_vector_type(8)))  float    v8f;
typedef unsigned int u32x4 __attribute__((ext_vector_type(4)));
typedef int          i32x8 __attribute__((ext_vector_type(8)));
typedef int          i32x4 __attribute__((ext_vector_type(4)));

#define H_     64
#define P_     8192
#define N_     8192
#define ALPHA_ 0.2f

#ifndef __has_builtin
#define __has_builtin(x) 0
#endif

#if __has_builtin(__builtin_amdgcn_tensor_load_to_lds)
#define HAVE_TDM 1
#else
#define HAVE_TDM 0
#endif

#if defined(__has_include)
#if __has_include(<hip/amd_detail/amd_gfx1250_TDM.h>)
#define TDM_6ARG 1
#else
#define TDM_6ARG 0
#endif
#else
#define TDM_6ARG 0
#endif

#if __has_builtin(__builtin_amdgcn_s_wait_tensorcnt)
#define WAIT_TENSORCNT(n) \
  do { __builtin_amdgcn_s_wait_tensorcnt(n); asm volatile("" ::: "memory"); } while (0)
#else
#define WAIT_TENSORCNT(n) asm volatile("s_wait_tensorcnt " #n ::: "memory")
#endif

__device__ __forceinline__ float sigmoidf_(float v) { return 1.f / (1.f + __expf(-v)); }
__device__ __forceinline__ float eluf_(float v)     { return v > 0.f ? v : (__expf(v) - 1.f); }

__device__ __forceinline__ unsigned lds_off_of(const void* p) {
  // generic pointers to LDS carry the byte offset in the low 32 bits
  return (unsigned)(uintptr_t)p;
}

__device__ __forceinline__ void unpack4(float* dst, float4 v) {
  dst[0] = v.x; dst[1] = v.y; dst[2] = v.z; dst[3] = v.w;
}

#if HAVE_TDM
// Issue a TDM DMA of a [tile_rows x tile_cols] fp32 tile (row stride
// stride_elems) from global memory into LDS at byte offset lds_off.
// D# layout per CDNA5 ISA ch.8.
__device__ __forceinline__ void tdm_load_2d_f32(const float* g, unsigned lds_off,
                                                int tile_cols, int tile_rows,
                                                int stride_elems) {
  const unsigned long long ga = (unsigned long long)(uintptr_t)g;
  u32x4 g0;
  g0[0] = 1u;                                             // count=1, user mode
  g0[1] = lds_off;                                        // lds_addr (bytes)
  g0[2] = (unsigned)(ga & 0xffffffffull);                 // global_addr[31:0]
  g0[3] = (unsigned)((ga >> 32) & 0x1ffffffull) | (2u << 30);  // ga[56:32] | type=2
  i32x8 g1;
  g1[0] = 0x00020000;                                     // data_size = 4B
  g1[1] = (int)(((unsigned)stride_elems & 0xffffu) << 16);          // tensor_dim0 lo
  g1[2] = (int)((((unsigned)stride_elems >> 16) & 0xffffu)
              | (((unsigned)N_ & 0xffffu) << 16));                  // dim0 hi | dim1 lo
  g1[3] = (int)((((unsigned)N_ >> 16) & 0xffffu)
              | (((unsigned)tile_cols & 0xffffu) << 16));           // dim1 hi | tile_dim0
  g1[4] = (int)((unsigned)tile_rows & 0xffffu);                     // tile_dim1 (dim2=0)
  g1[5] = (int)(unsigned)stride_elems;                              // dim0_stride[31:0]
  g1[6] = 0;
  g1[7] = 0;
  i32x4 z4 = {0, 0, 0, 0};
#if TDM_6ARG
  i32x8 z8 = {0, 0, 0, 0, 0, 0, 0, 0};
  __builtin_amdgcn_tensor_load_to_lds(g0, g1, z4, z4, z8, 0);
#else
  __builtin_amdgcn_tensor_load_to_lds(g0, g1, z4, z4, 0);
#endif
}
#endif  // HAVE_TDM

// ----------------------------- GRU cells -----------------------------------
__global__ void k_gru(const float* __restrict__ x, const float* __restrict__ user,
                      const float* __restrict__ h1old, const float* __restrict__ h3old,
                      const float* __restrict__ Wih1, const float* __restrict__ Whh1,
                      const float* __restrict__ bih1, const float* __restrict__ bhh1,
                      const float* __restrict__ Wih3, const float* __restrict__ Whh3,
                      const float* __restrict__ bih3, const float* __restrict__ bhh3,
                      float* nh, float* ph, float* out_nh, float* out_ph) {
  __shared__ float xin[9];
  __shared__ float hs[64];
  const int t = threadIdx.x;
  const int which = blockIdx.x;           // 0: seq GRU, 1: period GRU
  const float* hold = which ? h3old : h1old;
  const float* Wih  = which ? Wih3 : Wih1;
  const float* Whh  = which ? Whh3 : Whh1;
  const float* bih  = which ? bih3 : bih1;
  const float* bhh  = which ? bhh3 : bhh1;
  const int IN2 = which ? 9 : 8;
  if (t < 8) xin[t + which] = x[t];
  if (which && t == 0) xin[0] = user[0];
  hs[t] = hold[t];
  __syncthreads();
  float gi[3], gh[3];
#pragma unroll
  for (int g = 0; g < 3; ++g) {
    const int rI = g * 64 + t;
    float si = bih[rI];
    for (int k = 0; k < IN2; ++k) si += xin[k] * Wih[rI * IN2 + k];
    gi[g] = si;
    float sh = bhh[rI];
    for (int k = 0; k < 64; ++k) sh += hs[k] * Whh[rI * 64 + k];
    gh[g] = sh;
  }
  const float r = sigmoidf_(gi[0] + gh[0]);
  const float z = sigmoidf_(gi[1] + gh[1]);
  const float n = tanhf(gi[2] + r * gh[2]);
  const float hnew = (1.f - z) * n + z * hs[t];
  if (which) { ph[t] = hnew; out_ph[t] = hnew; }
  else       { nh[t] = hnew; out_nh[t] = hnew; }
}

// ------------------------- POI distance attention ---------------------------
__global__ void k_poi1(const float* __restrict__ nh, const float* __restrict__ loc,
                       const float* __restrict__ dist, const float* __restrict__ w,
                       float* allwe) {
  const int h = threadIdx.x & 63, e = threadIdx.x >> 6;  // 128 threads
  const float wh = w[0] * nh[h];
  float part = 0.f;
  const int p0 = blockIdx.x * 128;
  for (int p = p0; p < p0 + 128; ++p) {
    const float dp = __expf(-dist[p] * 0.5f);
    part += __expf(wh * loc[2 * p + e] * dp);
  }
  atomicAdd(&allwe[h * 2 + e], part);
}

__global__ void k_poi2(const float* __restrict__ nh, const float* __restrict__ loc,
                       const float* __restrict__ dist, const float* __restrict__ w,
                       const float* __restrict__ allwe, float* cL) {
  const int h = threadIdx.x & 63, e = threadIdx.x >> 6;
  const float wh = w[0] * nh[h];
  const float inv_aw = 1.f / allwe[h * 2 + e];
  float part = 0.f;
  const int p0 = blockIdx.x * 128;
  for (int p = p0; p < p0 + 128; ++p) {
    const float dp = __expf(-dist[p] * 0.5f);
    const float lp = loc[2 * p + e];
    part += __expf(wh * lp * dp) * inv_aw * lp;
  }
  atomicAdd(&cL[h], part);
}

// --------------------- periodicity update + cP ------------------------------
__global__ void k_period(const float* __restrict__ nh, const float* __restrict__ ph,
                         const float* __restrict__ qhh, const float* __restrict__ aH,
                         float* out_qhh, float* out_aH, float* cP) {
  __shared__ float red[64];
  const int a = blockIdx.x, b = threadIdx.x;
  const float qhi = __expf(nh[a] * ph[b]);
  const float qn  = qhh[a * 64 + b] + qhi;
  const float an  = aH[a * 64 + b] + qhi / qn;
  out_qhh[a * 64 + b] = qn;
  out_aH[a * 64 + b]  = an;
  red[b] = an * ph[b];
  __syncthreads();
  for (int s = 32; s; s >>= 1) { if (b < s) red[b] += red[b + s]; __syncthreads(); }
  if (!b) cP[a] = red[0];
}

// --------------------------- GAT prep (layer 1) -----------------------------
// WhCM: column-major fp16 copy of Wh, [head*4+f][N] -> B fragments become two
// contiguous 16B runs per lane (k = hi*8+{0..7}, +16).
__global__ void k_prep1(const float* __restrict__ nodes, const float* __restrict__ gatW,
                        const float* __restrict__ gatA,
                        _Float16* WhCM, float* e1, float* e2) {
  const int n = blockIdx.x * blockDim.x + threadIdx.x;
  if (n >= N_) return;
  float nd[4];
#pragma unroll
  for (int c = 0; c < 4; ++c) nd[c] = nodes[n * 4 + c];
#pragma unroll
  for (int k = 0; k < 4; ++k) {
    float Wh[4];
#pragma unroll
    for (int f = 0; f < 4; ++f) {
      float s = 0.f;
#pragma unroll
      for (int c = 0; c < 4; ++c) s += nd[c] * gatW[k * 16 + c * 4 + f];
      Wh[f] = s;
      WhCM[(size_t)(k * 4 + f) * N_ + n] = (_Float16)s;
    }
    float s1 = 0.f, s2 = 0.f;
#pragma unroll
    for (int f = 0; f < 4; ++f) { s1 += Wh[f] * gatA[k * 8 + f]; s2 += Wh[f] * gatA[k * 8 + 4 + f]; }
    e1[k * N_ + n] = s1;
    e2[k * N_ + n] = s2;
  }
}

// ------------------ fused 4-head flash GAT (layer 1) ------------------------
// 512 blocks x 128 threads. Wave w = head w; adj tile shared by all heads,
// DMA'd by wave 0 via TDM with double buffering. No softmax running max:
// logits are 0.1-scale and masked probs are exactly 0.
__global__ void k_gat1(const float* __restrict__ adj, const float* __restrict__ e1,
                       const float* __restrict__ e2, const _Float16* __restrict__ WhCM,
                       float* __restrict__ x1) {
#if HAVE_TDM
  __shared__ __align__(16) float adjT[2][16 * 32];
#else
  __shared__ __align__(16) float adjT[1][16 * 32];
#endif
  const int wave = threadIdx.x >> 5;      // head index
  const int lane = threadIdx.x & 31;
  const int i0   = blockIdx.x * 16;
  const int row  = lane & 15;
  const int hi   = lane >> 4;
  const int col  = lane & 15;
  const float e1row = e1[wave * N_ + i0 + row];
  const float* e2base = e2 + wave * N_;
  float l = 0.f;
  v8f acc = {};
  const int NT = N_ / 32;
#if HAVE_TDM
  if (wave == 0)
    tdm_load_2d_f32(adj + (size_t)i0 * N_, lds_off_of(&adjT[0][0]), 32, 16, N_);
#endif
  for (int jt = 0; jt < NT; ++jt) {
    const int j0 = jt * 32;
#if HAVE_TDM
    const float* curAdj = &adjT[jt & 1][0];
    if (wave == 0) {
      if (jt + 1 < NT) {   // prefetch next tile into the other buffer
        tdm_load_2d_f32(adj + (size_t)i0 * N_ + (size_t)(jt + 1) * 32,
                        lds_off_of(&adjT[(jt + 1) & 1][0]), 32, 16, N_);
        WAIT_TENSORCNT(1); // current tile complete, prefetch in flight
      } else {
        WAIT_TENSORCNT(0);
      }
    }
#else
    const float* curAdj = &adjT[0][0];
    {  // 128 threads x one b128 = 2KB tile
      const int r = threadIdx.x >> 3, c = (threadIdx.x & 7) * 4;
      *(float4*)&adjT[0][r * 32 + c] =
          *(const float4*)(adj + (size_t)(i0 + r) * N_ + j0 + c);
    }
#endif
    __syncthreads();
    // vectorized operand fetch: two 8-element runs (K = hi*8+{0..7}, +16)
    float adjv[16], e2v[16];
    {
      const float* ar = curAdj + row * 32 + hi * 8;      // LDS, 4x b128
      unpack4(adjv + 0,  *(const float4*)(ar + 0));
      unpack4(adjv + 4,  *(const float4*)(ar + 4));
      unpack4(adjv + 8,  *(const float4*)(ar + 16));
      unpack4(adjv + 12, *(const float4*)(ar + 20));
      const float* er = e2base + j0 + hi * 8;            // global, 4x b128
      unpack4(e2v + 0,  *(const float4*)(er + 0));
      unpack4(e2v + 4,  *(const float4*)(er + 4));
      unpack4(e2v + 8,  *(const float4*)(er + 16));
      unpack4(e2v + 12, *(const float4*)(er + 20));
    }
    // A fragment: branchless masked exp
    v16h a;
#pragma unroll
    for (int i = 0; i < 16; ++i) {
      float e = e1row + e2v[i];
      e = e > 0.f ? e : ALPHA_ * e;                       // leaky relu
      const float pe = __expf(e);
      const float p  = adjv[i] > 0.f ? pe : 0.f;
      l += p;
      a[i] = (_Float16)p;
    }
    // B fragment: two 8-half runs of column-major Wh (K = hi*8+{0..7}, +16)
    v16h b = {};
    if (col < 4) {
      const _Float16* br = WhCM + (size_t)(wave * 4 + col) * N_ + j0 + hi * 8;
#pragma unroll
      for (int s = 0; s < 16; ++s) b[s] = br[(s < 8) ? s : (s + 8)];
    }
    acc = __builtin_amdgcn_wmma_f32_16x16x32_f16(false, a, false, b, (short)0, acc,
                                                 false, false);
    __syncthreads();  // tile buffer fully consumed before next overwrite
  }
  const float ltot = l + __shfl_xor(l, 16, 32);
#pragma unroll
  for (int r = 0; r < 8; ++r) {
    const int orow = r + (hi ? 8 : 0);
    const float lr = __shfl(ltot, orow, 32);
    if (col < 4) {
      float v = acc[r] / lr;
      x1[(size_t)(i0 + orow) * 16 + wave * 4 + col] = eluf_(v);
    }
  }
}

// --------------------------- GAT prep (layer 2) -----------------------------
__global__ void k_prep2(const float* __restrict__ x1, const float* __restrict__ Wout,
                        const float* __restrict__ aout,
                        _Float16* WhoCM, float* e1o, float* e2o) {
  const int n = blockIdx.x * blockDim.x + threadIdx.x;
  if (n >= N_) return;
  float Who[2] = {0.f, 0.f};
#pragma unroll
  for (int c = 0; c < 16; ++c) {
    const float xv = x1[(size_t)n * 16 + c];
    Who[0] += xv * Wout[c * 2 + 0];
    Who[1] += xv * Wout[c * 2 + 1];
  }
  WhoCM[(size_t)0 * N_ + n] = (_Float16)Who[0];   // column-major [f][N]
  WhoCM[(size_t)1 * N_ + n] = (_Float16)Who[1];
  e1o[n] = Who[0] * aout[0] + Who[1] * aout[1];
  e2o[n] = Who[0] * aout[2] + Who[1] * aout[3];
}

// --------------------- flash GAT output layer (F'=2) ------------------------
// Waves split the j-range; each wave DMAs its OWN adj tiles (double-buffered
// TDM, no workgroup barrier in the inner loop). Partial sums merge through LDS
// at the end (valid: no softmax rescaling). Epilogue: elu + log_softmax.
__global__ void k_gat2(const float* __restrict__ adj, const float* __restrict__ e1o,
                       const float* __restrict__ e2o, const _Float16* __restrict__ WhoCM,
                       float* __restrict__ out2) {
#if HAVE_TDM
  __shared__ __align__(16) float adjT[4][2][16 * 32];
#else
  __shared__ __align__(16) float adjT[4][1][16 * 32];
#endif
  __shared__ float accS[4][16][16];
  __shared__ float lS[4][16];
  const int wave = threadIdx.x >> 5;
  const int lane = threadIdx.x & 31;
  const int i0   = blockIdx.x * 16;
  const int row  = lane & 15;
  const int hi   = lane >> 4;
  const int col  = lane & 15;
  const float e1row = e1o[i0 + row];
  float l = 0.f;
  v8f acc = {};
  const int NT = N_ / 128;   // tiles per wave
#if HAVE_TDM
  tdm_load_2d_f32(adj + (size_t)i0 * N_ + wave * 32,
                  lds_off_of(&adjT[wave][0][0]), 32, 16, N_);
#endif
  for (int jt = 0; jt < NT; ++jt) {
    const int j0 = wave * 32 + jt * 128;
#if HAVE_TDM
    const float* curAdj = &adjT[wave][jt & 1][0];
    if (jt + 1 < NT) {
      tdm_load_2d_f32(adj + (size_t)i0 * N_ + (size_t)(j0 + 128),
                      lds_off_of(&adjT[wave][(jt + 1) & 1][0]), 32, 16, N_);
      WAIT_TENSORCNT(1);
    } else {
      WAIT_TENSORCNT(0);
    }
#else
    const float* curAdj = &adjT[wave][0][0];
#pragma unroll
    for (int s = 0; s < 4; ++s) {   // 32 lanes x 4 b128 = 2KB tile
      const int q = lane * 16 + s * 4;
      const int r = q >> 5, c = q & 31;
      *(float4*)&adjT[wave][0][q] =
          *(const float4*)(adj + (size_t)(i0 + r) * N_ + j0 + c);
    }
#endif
    float adjv[16], e2v[16];
    {
      const float* ar = curAdj + row * 32 + hi * 8;
      unpack4(adjv + 0,  *(const float4*)(ar + 0));
      unpack4(adjv + 4,  *(const float4*)(ar + 4));
      unpack4(adjv + 8,  *(const float4*)(ar + 16));
      unpack4(adjv + 12, *(const float4*)(ar + 20));
      const float* er = e2o + j0 + hi * 8;
      unpack4(e2v + 0,  *(const float4*)(er + 0));
      unpack4(e2v + 4,  *(const float4*)(er + 4));
      unpack4(e2v + 8,  *(const float4*)(er + 16));
      unpack4(e2v + 12, *(const float4*)(er + 20));
    }
    v16h a;
#pragma unroll
    for (int i = 0; i < 16; ++i) {
      float e = e1row + e2v[i];
      e = e > 0.f ? e : ALPHA_ * e;
      const float pe = __expf(e);
      const float p  = adjv[i] > 0.f ? pe : 0.f;
      l += p;
      a[i] = (_Float16)p;
    }
    v16h b = {};
    if (col < 2) {
      const _Float16* br = WhoCM + (size_t)col * N_ + j0 + hi * 8;
#pragma unroll
      for (int s = 0; s < 16; ++s) b[s] = br[(s < 8) ? s : (s + 8)];
    }
    acc = __builtin_amdgcn_wmma_f32_16x16x32_f16(false, a, false, b, (short)0, acc,
                                                 false, false);
  }
  const float ltot = l + __shfl_xor(l, 16, 32);
  if (lane < 16) lS[wave][lane] = ltot;
#pragma unroll
  for (int r = 0; r < 8; ++r) accS[wave][r + (hi ? 8 : 0)][col] = acc[r];
  __syncthreads();
  if (wave == 0) {
#pragma unroll
    for (int r = 0; r < 8; ++r) {
      const int orow = r + (hi ? 8 : 0);
      const float s  = accS[0][orow][col] + accS[1][orow][col] +
                       accS[2][orow][col] + accS[3][orow][col];
      const float lr = lS[0][orow] + lS[1][orow] + lS[2][orow] + lS[3][orow];
      const float v  = eluf_(s / lr);
      const float o  = __shfl_xor(v, 1, 32);       // partner column (0<->1)
      const float m  = fmaxf(v, o);
      const float lse = m + __logf(__expf(v - m) + __expf(o - m));
      if (col < 2) out2[(size_t)(i0 + orow) * 2 + col] = v - lse;
    }
  }
}

// ------------------------ mL range, logits, top-5 ---------------------------
__global__ void k_mlrange(const float* __restrict__ nh, const float* __restrict__ cL,
                          const float* __restrict__ cP, float* mlmm) {
  __shared__ float smax[64], smin[64];
  const int t = threadIdx.x;
  const float a = nh[t], b = cL[t], c = cP[t];
  smax[t] = fmaxf(a, fmaxf(b, c));
  smin[t] = fminf(a, fminf(b, c));
  __syncthreads();
  for (int s = 32; s; s >>= 1) {
    if (t < s) { smax[t] = fmaxf(smax[t], smax[t + s]); smin[t] = fminf(smin[t], smin[t + s]); }
    __syncthreads();
  }
  if (!t) { mlmm[0] = smax[0]; mlmm[1] = smin[0]; }
}

__global__ void k_logits(const float* __restrict__ out2, const float* __restrict__ loc,
                         const int* __restrict__ pre, const float* __restrict__ mlmm,
                         float* logits) {
  const int p = blockIdx.x * blockDim.x + threadIdx.x;
  if (p >= P_) return;
  int off = pre[0];
  if (off < 0) off = 0;
  if (off > N_ - P_) off = N_ - P_;
  const float g0 = out2[(size_t)(off + p) * 2 + 0];
  const float g1 = out2[(size_t)(off + p) * 2 + 1];
  const float s  = g0 * loc[2 * p] + g1 * loc[2 * p + 1];
  logits[p] = (s >= 0.f) ? s * mlmm[0] : s * mlmm[1];  // max_j s*mL_j
}

__global__ void k_top5(const float* __restrict__ logits, float* out_idx) {
  const int lane = threadIdx.x;
  int found[5];
  for (int t = 0; t < 5; ++t) {
    float bm = -3.4e38f;
    int bi = 0x7fffffff;
    for (int p = lane; p < P_; p += 32) {
      bool skip = false;
      for (int q = 0; q < t; ++q) skip |= (found[q] == p);
      if (skip) continue;
      const float v = logits[p];
      if (v > bm || (v == bm && p < bi)) { bm = v; bi = p; }
    }
    for (int off = 16; off; off >>= 1) {
      const float om = __shfl_xor(bm, off, 32);
      const int   oi = __shfl_xor(bi, off, 32);
      if (om > bm || (om == bm && oi < bi)) { bm = om; bi = oi; }
    }
    found[t] = bi;
  }
  if (lane == 0)
    for (int t = 0; t < 5; ++t) out_idx[t] = (float)found[t];
}

// ---------------------------------------------------------------------------
extern "C" void kernel_launch(void* const* d_in, const int* in_sizes, int n_in,
                              void* d_out, int out_size, void* d_ws, size_t ws_size,
                              hipStream_t stream) {
  const float* x        = (const float*)d_in[0];
  const float* user     = (const float*)d_in[1];
  const float* nextHid  = (const float*)d_in[2];
  const float* periodHid= (const float*)d_in[3];
  const float* qhh      = (const float*)d_in[4];
  const float* aH       = (const float*)d_in[5];
  const float* poi_loc  = (const float*)d_in[6];
  const float* poi_dist = (const float*)d_in[7];
  const float* nodes    = (const float*)d_in[8];
  const float* edges    = (const float*)d_in[9];
  const int*   pre      = (const int*)  d_in[10];
  const float* w        = (const float*)d_in[11];
  const float* Wih1     = (const float*)d_in[12];
  const float* Whh1     = (const float*)d_in[13];
  const float* bih1     = (const float*)d_in[14];
  const float* bhh1     = (const float*)d_in[15];
  const float* Wih3     = (const float*)d_in[16];
  const float* Whh3     = (const float*)d_in[17];
  const float* bih3     = (const float*)d_in[18];
  const float* bhh3     = (const float*)d_in[19];
  const float* gatW     = (const float*)d_in[20];
  const float* gatA     = (const float*)d_in[21];
  const float* gatWout  = (const float*)d_in[22];
  const float* gatAout  = (const float*)d_in[23];

  float* out = (float*)d_out;
  float* out_nh   = out;            // [64]
  float* out_ph   = out + 64;       // [64]
  float* out_qhh  = out + 128;      // [4096]
  float* out_aH   = out + 4224;     // [4096]
  float* out_idx  = out + 8320;     // [5]

  // workspace layout (float offsets); ~1.25 MB total
  float* wsf    = (float*)d_ws;
  float* nh     = wsf + 0;                    // [64]
  float* ph     = wsf + 64;                   // [64]
  float* cL     = wsf + 128;                  // [64]  (atomics)
  float* cP     = wsf + 192;                  // [64]
  float* allwe  = wsf + 256;                  // [128] (atomics)
  float* mlmm   = wsf + 384;                  // [2]
  float* e1     = wsf + 512;                  // [4*8192]
  float* e2     = wsf + 33280;                // [4*8192]
  float* e1o    = wsf + 66048;                // [8192]
  float* e2o    = wsf + 74240;                // [8192]
  float* x1     = wsf + 82432;                // [8192*16]
  float* out2   = wsf + 213504;               // [8192*2]
  float* logits = wsf + 229888;               // [8192]
  _Float16* WhCM  = (_Float16*)(wsf + 238080); // [16][8192] halfs (col-major)
  _Float16* WhoCM = (_Float16*)(wsf + 303616); // [2][8192] halfs (col-major)

  (void)in_sizes; (void)n_in; (void)out_size; (void)ws_size;

  // zero the atomic accumulators (cL, cP, allwe)
  hipMemsetAsync((void*)(wsf + 128), 0, 256 * sizeof(float), stream);

  k_gru<<<2, 64, 0, stream>>>(x, user, nextHid, periodHid,
                              Wih1, Whh1, bih1, bhh1,
                              Wih3, Whh3, bih3, bhh3,
                              nh, ph, out_nh, out_ph);
  k_poi1<<<64, 128, 0, stream>>>(nh, poi_loc, poi_dist, w, allwe);
  k_poi2<<<64, 128, 0, stream>>>(nh, poi_loc, poi_dist, w, allwe, cL);
  k_period<<<64, 64, 0, stream>>>(nh, ph, qhh, aH, out_qhh, out_aH, cP);

  k_prep1<<<32, 256, 0, stream>>>(nodes, gatW, gatA, WhCM, e1, e2);
  k_gat1<<<N_ / 16, 128, 0, stream>>>(edges, e1, e2, WhCM, x1);
  k_prep2<<<32, 256, 0, stream>>>(x1, gatWout, gatAout, WhoCM, e1o, e2o);
  k_gat2<<<N_ / 16, 128, 0, stream>>>(edges, e1o, e2o, WhoCM, out2);

  k_mlrange<<<1, 64, 0, stream>>>(nh, cL, cP, mlmm);
  k_logits<<<32, 256, 0, stream>>>(out2, poi_loc, pre, mlmm, logits);
  k_top5<<<1, 32, 0, stream>>>(logits, out_idx);
}
